// MultiheadAttention_81475529605383
// MI455X (gfx1250) — matrix operand order repaired and checked by reading, not compiled, not run on hardware
//
#include <hip/hip_runtime.h>

// ---------------------------------------------------------------------------
// MHA forward for MI455X (gfx1250, wave32, WMMA bf16 16x16x32, f32 accum).
// Compute-bound (~137 GFLOP vs ~9us of HBM traffic at 23.3 TB/s):
//  - all GEMMs + attention run on v_wmma_f32_16x16x32_bf16
//  - operand tiles staged in LDS with GLOBAL_LOAD_ASYNC_TO_LDS_B128
//    (ASYNCcnt), double-buffered; pipeline unrolled by the buffer period so
//    tile k+1 copies overlap WMMAs on tile k with no buffer-index math
//  - flash attention (online softmax), scores never materialized; K/V tiles
//    shared by all 4 waves of a block through LDS
// ---------------------------------------------------------------------------

#define B_DIM 4
#define S_DIM 2048
#define H_DIM 16
#define D_DIM 64
#define DM    1024
#define MTOT  (B_DIM * S_DIM)   // 8192

typedef __attribute__((ext_vector_type(16))) __bf16           v16bf;
typedef __attribute__((ext_vector_type(8)))  float            v8f;
typedef __attribute__((ext_vector_type(8)))  unsigned short   v8u16;
typedef __attribute__((ext_vector_type(4)))  unsigned short   v4u16;

union FragBF {            // 16x32 A-frag or 32x16 B-frag, 8 VGPRs of bf16 pairs
    v16bf f;
    v8u16 h[2];           // h[0] -> VGPRs 0-3, h[1] -> VGPRs 4-7
};

// Async copy 16B global -> LDS (per lane). Tracked by ASYNCcnt.
__device__ __forceinline__ void async_b128(unsigned lds_off,
                                           const unsigned short* g) {
    asm volatile("global_load_async_to_lds_b128 %0, %1, off"
                 :: "v"(lds_off), "v"(g) : "memory");
}
#define ASYNC_WAIT4() asm volatile("s_wait_asynccnt 0x4" ::: "memory")
#define ASYNC_WAIT0() asm volatile("s_wait_asynccnt 0x0" ::: "memory")

__device__ __forceinline__ unsigned lds_addr(const unsigned short* p) {
    return (unsigned)(size_t)p;   // flat->LDS: low 32 bits (ISA 10.2)
}

__device__ __forceinline__ unsigned short f32_to_bf16(float x) {
    unsigned int u = __float_as_uint(x);
    unsigned int r = u + 0x7FFFu + ((u >> 16) & 1u);   // round-to-nearest-even
    return (unsigned short)(r >> 16);
}

__device__ __forceinline__ v8f zero8() {
    v8f z;
#pragma unroll
    for (int r = 0; r < 8; ++r) z[r] = 0.0f;
    return z;
}

// A-matrix 16x32 bf16 (ISA 7.12.2): lanes 0-15 M=0..15 K-halves {0..7,16..23},
// lanes 16-31 same M, K-halves {8..15,24..31}. Two B128 loads per lane.
__device__ __forceinline__ FragBF load_fragA_g(const unsigned short* p, int row0,
                                               int ld, int k0, int lane) {
    const int m = lane & 15, half = lane >> 4;
    const unsigned short* r = p + (size_t)(row0 + m) * ld + k0;
    FragBF f;
    f.h[0] = *(const v8u16*)(r + half * 8);
    f.h[1] = *(const v8u16*)(r + 16 + half * 8);
    return f;
}

__device__ __forceinline__ v8f wmma_bf16(const FragBF& a, const FragBF& b, v8f c) {
    return __builtin_amdgcn_wmma_f32_16x16x32_bf16(
        false, a.f, false, b.f, (short)0, c, false, false);
}

// ---------------------------------------------------------------------------
// f32 -> bf16 bit conversion, float4 loads, 8B stores.
// ---------------------------------------------------------------------------
__global__ __launch_bounds__(256)
void cvt_f32_bf16(const float* __restrict__ in, unsigned short* __restrict__ out,
                  int n4) {
    int i = blockIdx.x * 256 + threadIdx.x;
    if (i >= n4) return;
    float4 v = ((const float4*)in)[i];
    v4u16 o;
    o[0] = f32_to_bf16(v.x); o[1] = f32_to_bf16(v.y);
    o[2] = f32_to_bf16(v.z); o[3] = f32_to_bf16(v.w);
    ((v4u16*)out)[i] = o;
}

// ---------------------------------------------------------------------------
// GEMM: C[M,N] = A[M,K] @ W[N,K]^T, 256 thr = 8 waves, block tile 128x128,
// wave tile 64x32 (8 WMMA / k-step). A and B k-slices (128x32 bf16) staged in
// LDS by GLOBAL_LOAD_ASYNC_TO_LDS_B128, double buffered, loop unrolled by 2
// (one body per buffer). Row pitch 40 elems (80B, 16B-aligned, conflict-free
// for 16-row x 16B fragment reads).
// MODE 0: bf16 out -> Q/K layout [BH,S,64]; MODE 1: bf16 out -> Vt [BH,D,S]
// (B128 packed stores, 8 consecutive s per lane); MODE 2: f32 out, plain [M,N].
// ---------------------------------------------------------------------------
#define APITCH 40
#define KT_GEMM (DM / 32)    // 32 k-tiles
template <int MODE>
__global__ __launch_bounds__(256)
void gemm_bf16(const unsigned short* __restrict__ A,
               const unsigned short* __restrict__ W,
               void* __restrict__ Out) {
    __shared__ unsigned short ldsA[2][128 * APITCH];
    __shared__ unsigned short ldsB[2][128 * APITCH];
    const int tid   = threadIdx.x;
    const int lane  = tid & 31;
    const int wave  = tid >> 5;
    const int half  = lane >> 4;
    const int nlane = lane & 15;
    const int m0 = blockIdx.y * 128;
    const int n0 = blockIdx.x * 128;
    const int wm = (wave >> 2) * 64;     // wave M offset inside block tile
    const int wn = (wave & 3) * 32;      // wave N offset inside block tile

    // stage one 128x32 k-slice of A and of W: 4 async issues per thread total
    auto stage = [&](int k, unsigned short* bufA, unsigned short* bufB) {
#pragma unroll
        for (int it = 0; it < 2; ++it) {
            const int task = tid + it * 256;
            const int row = task >> 2, seg = task & 3;
            async_b128(lds_addr(bufA + row * APITCH + seg * 8),
                       A + (size_t)(m0 + row) * DM + k + seg * 8);
            async_b128(lds_addr(bufB + row * APITCH + seg * 8),
                       W + (size_t)(n0 + row) * DM + k + seg * 8);
        }
    };

    v8f acc[4][2];
#pragma unroll
    for (int i = 0; i < 4; ++i)
#pragma unroll
        for (int j = 0; j < 2; ++j) acc[i][j] = zero8();

    auto consume = [&](const unsigned short* la, const unsigned short* lb) {
        FragBF a[4], b[2];
#pragma unroll
        for (int i = 0; i < 4; ++i) {
            const int ra = wm + i * 16 + nlane;
            a[i].h[0] = *(const v8u16*)(la + ra * APITCH + half * 8);
            a[i].h[1] = *(const v8u16*)(la + ra * APITCH + 16 + half * 8);
        }
#pragma unroll
        for (int j = 0; j < 2; ++j) {
            const int rb = wn + j * 16 + nlane;
            b[j].h[0] = *(const v8u16*)(lb + rb * APITCH + half * 8);
            b[j].h[1] = *(const v8u16*)(lb + rb * APITCH + 16 + half * 8);
        }
#pragma unroll
        for (int i = 0; i < 4; ++i)
#pragma unroll
            for (int j = 0; j < 2; ++j)
                acc[i][j] = wmma_bf16(a[i], b[j], acc[i][j]);
    };

    stage(0,  ldsA[0], ldsB[0]);
    stage(32, ldsA[1], ldsB[1]);               // 8 copies in flight

    for (int kt = 0; kt < KT_GEMM; kt += 2) {
        ASYNC_WAIT4();                         // tile kt landed (kt+1 flying)
        __syncthreads();
        consume(ldsA[0], ldsB[0]);
        __syncthreads();                       // buf0 reads done
        if (kt + 2 < KT_GEMM) {
            stage((kt + 2) * 32, ldsA[0], ldsB[0]);
            ASYNC_WAIT4();                     // tile kt+1 landed (kt+2 flying)
        } else {
            ASYNC_WAIT0();                     // last pair: drain tile kt+1
        }
        __syncthreads();
        consume(ldsA[1], ldsB[1]);
        __syncthreads();                       // buf1 reads done
        if (kt + 3 < KT_GEMM)
            stage((kt + 3) * 32, ldsA[1], ldsB[1]);
    }

    if (MODE == 1) {
        // Vt [BH,D,S]: per lane d fixed, s = base + r -> 8 contiguous bf16
#pragma unroll
        for (int i = 0; i < 4; ++i)
#pragma unroll
            for (int j = 0; j < 2; ++j) {
                const int srow = m0 + wm + i * 16 + half * 8;   // mult of 8
                const int col  = n0 + wn + j * 16 + nlane;
                const int b = srow >> 11, s = srow & (S_DIM - 1);
                const int h = col >> 6,  d = col & (D_DIM - 1);
                v8u16 pk;
#pragma unroll
                for (int r = 0; r < 8; ++r) pk[r] = f32_to_bf16(acc[i][j][r]);
                *(v8u16*)((unsigned short*)Out +
                          ((size_t)(b * H_DIM + h) * D_DIM + d) * S_DIM + s) = pk;
            }
    } else {
#pragma unroll
        for (int i = 0; i < 4; ++i)
#pragma unroll
            for (int j = 0; j < 2; ++j)
#pragma unroll
                for (int r = 0; r < 8; ++r) {
                    const int row = m0 + wm + i * 16 + r + half * 8; // M=b*S+s
                    const int col = n0 + wn + j * 16 + nlane;        // N=h*64+d
                    const float v = acc[i][j][r];
                    if (MODE == 2) {
                        ((float*)Out)[(size_t)row * DM + col] = v;
                    } else {
                        const int b = row >> 11, s = row & (S_DIM - 1);
                        const int h = col >> 6,  d = col & (D_DIM - 1);
                        ((unsigned short*)Out)
                            [((size_t)(b * H_DIM + h) * S_DIM + s) * D_DIM + d] =
                            f32_to_bf16(v);
                    }
                }
    }
}

// ---------------------------------------------------------------------------
// Flash attention. Q,K: [BH,S,64] bf16. Vt: [BH,64,S] bf16 (pre-transposed so
// PV B-fragments are key-contiguous). 4 waves/block share one head: the 32-key
// K tile (32x64) and V tile (64x32) are async-staged into LDS once per block
// and consumed by all waves (4x traffic reduction). P(16x32) goes C-layout ->
// per-wave LDS -> A-layout (per-wave LDS is in-order, no barrier needed).
// K pitch 72 elems (144B), V pitch 40 elems: 16B-aligned, conflict-free.
// Double-buffered, loop unrolled by the buffer period.
// ---------------------------------------------------------------------------
#define KPITCH 72
#define VPITCH 40
#define KT_ATT (S_DIM / 32)   // 64 key blocks
__global__ __launch_bounds__(128)
void flash_attn(const unsigned short* __restrict__ Q,
                const unsigned short* __restrict__ Km,
                const unsigned short* __restrict__ Vt,
                unsigned short* __restrict__ Ob) {
    __shared__ unsigned short ldsK[2][32 * KPITCH];
    __shared__ unsigned short ldsV[2][64 * VPITCH];
    __shared__ unsigned short Plds[4][16 * 32];
    const int tid   = threadIdx.x;
    const int lane  = tid & 31;
    const int wave  = tid >> 5;
    const int half  = lane >> 4;
    const int nlane = lane & 15;
    const int bh = blockIdx.x;                      // 0..63
    const int q0 = blockIdx.y * 64 + wave * 16;     // q-tile base
    const unsigned short* Qb = Q  + (size_t)bh * S_DIM * D_DIM;
    const unsigned short* Kb = Km + (size_t)bh * S_DIM * D_DIM;
    const unsigned short* Vb = Vt + (size_t)bh * D_DIM * S_DIM;
    unsigned short* pl = Plds[wave];

    // stage K tile (32x64) + V tile (64x32): 4 async issues per thread
    auto stageKV = [&](int kb, unsigned short* bufK, unsigned short* bufV) {
#pragma unroll
        for (int it = 0; it < 2; ++it) {
            const int t0 = tid + it * 128;
            const int kr = t0 >> 3, ks = t0 & 7;    // K: 32 rows x 8 segs
            async_b128(lds_addr(bufK + kr * KPITCH + ks * 8),
                       Kb + (size_t)(kb + kr) * D_DIM + ks * 8);
            const int vr = t0 >> 2, vs = t0 & 3;    // V: 64 rows x 4 segs
            async_b128(lds_addr(bufV + vr * VPITCH + vs * 8),
                       Vb + (size_t)vr * S_DIM + kb + vs * 8);
        }
    };

    FragBF aq[2];
    aq[0] = load_fragA_g(Qb, q0, D_DIM, 0,  lane);
    aq[1] = load_fragA_g(Qb, q0, D_DIM, 32, lane);

    float mrow[8], lrow[8];
    v8f o[4];
#pragma unroll
    for (int t = 0; t < 4; ++t) o[t] = zero8();
#pragma unroll
    for (int r = 0; r < 8; ++r) { mrow[r] = -3.0e38f; lrow[r] = 0.0f; }

    auto consume = [&](const unsigned short* lk, const unsigned short* lv) {
        // ---- scores: S = (Q @ K^T) * 1/sqrt(64), two 16x16 tiles ----
        v8f s0 = zero8(), s1 = zero8();
#pragma unroll
        for (int dk = 0; dk < 2; ++dk) {
            FragBF bk0, bk1;
            const int c0 = dk * 32 + half * 8;
            bk0.h[0] = *(const v8u16*)(lk + nlane * KPITCH + c0);
            bk0.h[1] = *(const v8u16*)(lk + nlane * KPITCH + c0 + 16);
            bk1.h[0] = *(const v8u16*)(lk + (16 + nlane) * KPITCH + c0);
            bk1.h[1] = *(const v8u16*)(lk + (16 + nlane) * KPITCH + c0 + 16);
            s0 = wmma_bf16(aq[dk], bk0, s0);
            s1 = wmma_bf16(aq[dk], bk1, s1);
        }
        // ---- online softmax over this 32-key block ----
        float p0[8], p1[8], corr[8];
#pragma unroll
        for (int r = 0; r < 8; ++r) {
            const float a = s0[r] * 0.125f;
            const float b = s1[r] * 0.125f;
            float bm = fmaxf(a, b);
            bm = fmaxf(bm, __shfl_xor(bm, 1, 32));
            bm = fmaxf(bm, __shfl_xor(bm, 2, 32));
            bm = fmaxf(bm, __shfl_xor(bm, 4, 32));
            bm = fmaxf(bm, __shfl_xor(bm, 8, 32));
            const float mnew = fmaxf(mrow[r], bm);
            corr[r] = __expf(mrow[r] - mnew);
            mrow[r] = mnew;
            p0[r] = __expf(a - mnew);
            p1[r] = __expf(b - mnew);
            float rs = p0[r] + p1[r];
            rs += __shfl_xor(rs, 1, 32);
            rs += __shfl_xor(rs, 2, 32);
            rs += __shfl_xor(rs, 4, 32);
            rs += __shfl_xor(rs, 8, 32);
            lrow[r] = lrow[r] * corr[r] + rs;
        }
#pragma unroll
        for (int t = 0; t < 4; ++t)
#pragma unroll
            for (int r = 0; r < 8; ++r) o[t][r] *= corr[r];

        // ---- P: C-layout -> LDS (row-major 16x32 bf16) -> A-layout ----
#pragma unroll
        for (int r = 0; r < 8; ++r) {
            const int row = r + half * 8;
            pl[row * 32 + nlane]      = f32_to_bf16(p0[r]);
            pl[row * 32 + 16 + nlane] = f32_to_bf16(p1[r]);
        }
        FragBF pa;
        pa.h[0] = *(const v8u16*)(pl + nlane * 32 + half * 8);
        pa.h[1] = *(const v8u16*)(pl + nlane * 32 + 16 + half * 8);

        // ---- O += P @ V, 4 d-tiles of 16, V rows are d (key contiguous) ----
#pragma unroll
        for (int t = 0; t < 4; ++t) {
            FragBF vb;
            const unsigned short* vr = lv + (t * 16 + nlane) * VPITCH + half * 8;
            vb.h[0] = *(const v8u16*)(vr);
            vb.h[1] = *(const v8u16*)(vr + 16);
            o[t] = wmma_bf16(pa, vb, o[t]);
        }
    };

    stageKV(0,  ldsK[0], ldsV[0]);
    stageKV(32, ldsK[1], ldsV[1]);             // 8 copies in flight

    for (int kbt = 0; kbt < KT_ATT; kbt += 2) {
        ASYNC_WAIT4();                         // block kbt landed
        __syncthreads();
        consume(ldsK[0], ldsV[0]);
        __syncthreads();
        if (kbt + 2 < KT_ATT) {
            stageKV((kbt + 2) * 32, ldsK[0], ldsV[0]);
            ASYNC_WAIT4();                     // block kbt+1 landed
        } else {
            ASYNC_WAIT0();
        }
        __syncthreads();
        consume(ldsK[1], ldsV[1]);
        __syncthreads();
        if (kbt + 3 < KT_ATT)
            stageKV((kbt + 3) * 32, ldsK[1], ldsV[1]);
    }

    // ---- normalize + store bf16 attn-out as [B*S, 1024] ----
    const int b = bh >> 4, h = bh & 15;
#pragma unroll
    for (int t = 0; t < 4; ++t)
#pragma unroll
        for (int r = 0; r < 8; ++r) {
            const float v = o[t][r] / lrow[r];
            const int row = q0 + r + half * 8;
            const int col = h * 64 + t * 16 + nlane;
            Ob[((size_t)(b * S_DIM + row)) * DM + col] = f32_to_bf16(v);
        }
}

// ---------------------------------------------------------------------------
extern "C" void kernel_launch(void* const* d_in, const int* in_sizes, int n_in,
                              void* d_out, int out_size, void* d_ws, size_t ws_size,
                              hipStream_t stream) {
    const float* x  = (const float*)d_in[0];
    const float* wq = (const float*)d_in[1];
    const float* wk = (const float*)d_in[2];
    const float* wv = (const float*)d_in[3];
    const float* wf = (const float*)d_in[4];

    const size_t MB = 1024ull * 1024ull;
    char* ws = (char*)d_ws;
    unsigned short* xb   = (unsigned short*)(ws +  0 * MB);  // 16 MB
    unsigned short* wqb  = (unsigned short*)(ws + 16 * MB);  //  2 MB
    unsigned short* wkb  = (unsigned short*)(ws + 18 * MB);
    unsigned short* wvb  = (unsigned short*)(ws + 20 * MB);
    unsigned short* wfb  = (unsigned short*)(ws + 22 * MB);
    unsigned short* qb   = (unsigned short*)(ws + 24 * MB);  // 16 MB [BH,S,64]
    unsigned short* kbuf = (unsigned short*)(ws + 40 * MB);  // 16 MB [BH,S,64]
    unsigned short* vtb  = (unsigned short*)(ws + 56 * MB);  // 16 MB [BH,64,S]
    unsigned short* aob  = (unsigned short*)(ws + 72 * MB);  // 16 MB [B*S,1024]

    // 1) demote to bf16
    const int nx4 = (MTOT * DM) / 4;
    const int nw4 = (DM * DM) / 4;
    cvt_f32_bf16<<<(nx4 + 255) / 256, 256, 0, stream>>>(x,  xb,  nx4);
    cvt_f32_bf16<<<(nw4 + 255) / 256, 256, 0, stream>>>(wq, wqb, nw4);
    cvt_f32_bf16<<<(nw4 + 255) / 256, 256, 0, stream>>>(wk, wkb, nw4);
    cvt_f32_bf16<<<(nw4 + 255) / 256, 256, 0, stream>>>(wv, wvb, nw4);
    cvt_f32_bf16<<<(nw4 + 255) / 256, 256, 0, stream>>>(wf, wfb, nw4);

    // 2) projections (WMMA bf16 + async LDS staging)
    dim3 ggrid(DM / 128, MTOT / 128);
    gemm_bf16<0><<<ggrid, 256, 0, stream>>>(xb, wqb, qb);
    gemm_bf16<0><<<ggrid, 256, 0, stream>>>(xb, wkb, kbuf);
    gemm_bf16<1><<<ggrid, 256, 0, stream>>>(xb, wvb, vtb);   // V transposed

    // 3) flash attention: 64 heads x 32 q-tiles-of-64
    flash_attn<<<dim3(B_DIM * H_DIM, S_DIM / 64), 128, 0, stream>>>(qb, kbuf, vtb, aob);

    // 4) final projection, f32 out
    gemm_bf16<2><<<ggrid, 256, 0, stream>>>(aob, wfb, d_out);
}
